// DG_fc_1700807050148
// MI455X (gfx1250) — matrix operand order repaired and checked by reading, not compiled, not run on hardware
//
#include <hip/hip_runtime.h>
#include <hip/hip_bf16.h>

// ---------------- WMMA types / helpers (CDNA5, wave32) ----------------
typedef __attribute__((ext_vector_type(16))) __bf16 v16bf;
typedef __attribute__((ext_vector_type(8)))  float  v8f;
typedef __attribute__((ext_vector_type(4)))  unsigned int tdm_u32x4;
typedef __attribute__((ext_vector_type(8)))  int          tdm_i32x8;
typedef __attribute__((ext_vector_type(4)))  int          tdm_i32x4;

#define WMMA_BF16(a, b, c) \
  __builtin_amdgcn_wmma_f32_16x16x32_bf16(false, (a), false, (b), (short)0, (c), false, false)

// A fragment: 16x32 bf16 tile, row-major in LDS (lda elements per row).
// Contiguous K order used here (not ISA table order): lanes 0-15 rows 0-15 (K 0-15); lanes 16-31 same rows, K 16-31.
__device__ __forceinline__ void load_frag_a(const __bf16* sm, int lda, int mbase, int lane, v16bf& a) {
  int m  = mbase + (lane & 15);
  int kg = (lane >> 4) << 4;
  const __bf16* row = sm + (size_t)m * lda;
#pragma unroll
  for (int p = 0; p < 8; ++p) {
    int klo = (p << 1) + kg;
    a[2 * p]     = row[klo];
    a[2 * p + 1] = row[klo + 1];
  }
}

// B fragment: 32x16 bf16 tile stored K-contiguous per output column: smT[n][k].
// Lanes 0-15: K 0-15; lanes 16-31: K 16-31.  (16 contiguous bf16 -> ds_load_b128 x2)
__device__ __forceinline__ void load_frag_b(const __bf16* smT, int ldb, int nbase, int lane, v16bf& b) {
  int n  = nbase + (lane & 15);
  int kg = (lane >> 4) << 4;
  const __bf16* row = smT + (size_t)n * ldb + kg;
#pragma unroll
  for (int p = 0; p < 8; ++p) {
    b[2 * p]     = row[2 * p];
    b[2 * p + 1] = row[2 * p + 1];
  }
}

// ---------------- problem constants ----------------
#define BATCH 8
#define NNODE 62
#define SEQL  128
#define FIN   128
#define HDIM  256
#define NHEAD 8
#define DK    32
#define M1    63488          // B*N*L == B*L*N (multiple of 64)
#define NGRAPH 1024          // B*L
#define EL    ((size_t)M1 * HDIM)

// ---------------- generic bf16 GEMM:  C = A(MxK) @ W(KxN) + bias ----------------
// block tile 64x128, 8 waves in 4x2 grid; each wave: 16x64 strip = 4 WMMA tiles / K-step.
// A tile staged via Tensor Data Mover (async, TENSORcnt); B tile via uint4 loads.
#define GEMM_LDS (64 * 32 * 2 + 128 * 32 * 2)   // smA 4KB @0, smB 8KB @4096
__global__ __launch_bounds__(256) void gemm_bf16(
    const __bf16* __restrict__ A, const __bf16* __restrict__ W,
    const float* __restrict__ bias, float* __restrict__ outf,
    __bf16* __restrict__ outb, int M, int N, int K) {
  extern __shared__ unsigned char gsm[];
  __bf16* smA = (__bf16*)gsm;            // [m][k] 64x32, LDS byte offset 0 (dynamic base)
  __bf16* smB = (__bf16*)(gsm + 4096);   // [n][k] 128x32
  int tid = threadIdx.x, lane = tid & 31, wave = tid >> 5;
  int m0 = blockIdx.y * 64, n0 = blockIdx.x * 128;
  int wm = (wave >> 1) << 4;             // 0,16,32,48
  int wn = (wave & 1) << 6;              // 0,64
  v8f acc0 = {}, acc1 = {}, acc2 = {}, acc3 = {};
  for (int k0 = 0; k0 < K; k0 += 32) {
    // ---- A tile (64x32) via TDM: 2D tensor descriptor, dim0 = K (contiguous) ----
#if __has_builtin(__builtin_amdgcn_tensor_load_to_lds)
    if (wave == 0) {
      unsigned long long ga = (unsigned long long)(uintptr_t)(A + (size_t)m0 * K + k0);
      tdm_u32x4 g0;
      g0[0] = 1u;                                   // count=1, user descriptor
      g0[1] = 0u;                                   // lds_addr = 0 (smA)
      g0[2] = (unsigned)(ga & 0xffffffffull);
      g0[3] = (unsigned)((ga >> 32) & 0x1ffffffull) | 0x80000000u;  // addr[56:32] | type=2
      unsigned td0 = (unsigned)K, td1 = 64u, tile0 = 32u, tile1 = 64u, st0 = (unsigned)K;
      tdm_i32x8 g1;
      g1[0] = (int)(1u << 16);                      // data_size=1 (2 bytes), mask=0
      g1[1] = (int)((td0 & 0xffffu) << 16);         // tensor_dim0[15:0] -> bits 63:48
      g1[2] = (int)((td0 >> 16) | ((td1 & 0xffffu) << 16));  // dim0 hi | dim1 lo
      g1[3] = (int)((td1 >> 16) | (tile0 << 16));   // dim1 hi | tile_dim0
      g1[4] = (int)(tile1 & 0xffffu);               // tile_dim1, tile_dim2=0
      g1[5] = (int)st0;                             // tensor_dim0_stride[31:0]
      g1[6] = 0;                                    // stride hi, dim1_stride lo
      g1[7] = 0;
      tdm_i32x4 g2 = {0, 0, 0, 0}, g3 = {0, 0, 0, 0};
#if defined(__clang_major__) && (__clang_major__ >= 23)
      tdm_i32x8 g4 = {0, 0, 0, 0, 0, 0, 0, 0};
      __builtin_amdgcn_tensor_load_to_lds(g0, g1, g2, g3, g4, 0);
#else
      __builtin_amdgcn_tensor_load_to_lds(g0, g1, g2, g3, 0);
#endif
    }
#else
    for (int i = tid; i < 64 * 32; i += 256) {
      int m = i >> 5, k = i & 31;
      int gm = m0 + m; if (gm >= M) gm = M - 1;
      smA[i] = A[(size_t)gm * K + k0 + k];
    }
#endif
    // ---- B tile (32x128) -> smB[n][k], vectorized 16B loads ----
    for (int c = tid; c < 512; c += 256) {
      int k = c >> 4, nc = (c & 15) << 3;
      union { uint4 u; __bf16 h[8]; } t;
      t.u = *(const uint4*)(W + (size_t)(k0 + k) * N + n0 + nc);
#pragma unroll
      for (int e = 0; e < 8; ++e) smB[(nc + e) * 32 + k] = t.h[e];
    }
    if (k0 + 32 < K) __builtin_prefetch(W + (size_t)(k0 + 32) * N + n0, 0, 1);
#if __has_builtin(__builtin_amdgcn_tensor_load_to_lds)
    if (wave == 0) __builtin_amdgcn_s_wait_tensorcnt(0);
#endif
    __syncthreads();
    v16bf a, b;
    load_frag_a(smA, 32, wm, lane, a);
    load_frag_b(smB, 32, wn +  0, lane, b); acc0 = WMMA_BF16(a, b, acc0);
    load_frag_b(smB, 32, wn + 16, lane, b); acc1 = WMMA_BF16(a, b, acc1);
    load_frag_b(smB, 32, wn + 32, lane, b); acc2 = WMMA_BF16(a, b, acc2);
    load_frag_b(smB, 32, wn + 48, lane, b); acc3 = WMMA_BF16(a, b, acc3);
    __syncthreads();
  }
  int mb = m0 + wm + ((lane >> 4) << 3);
  v8f accs[4] = {acc0, acc1, acc2, acc3};
#pragma unroll
  for (int t = 0; t < 4; ++t) {
    int nn = n0 + wn + (t << 4) + (lane & 15);
    float bv = bias ? bias[nn] : 0.f;
#pragma unroll
    for (int r = 0; r < 8; ++r) {
      int mm = mb + r;
      if (mm < M) {
        float v = accs[t][r] + bv;
        if (outf) outf[(size_t)mm * N + nn] = v;
        if (outb) outb[(size_t)mm * N + nn] = (__bf16)v;
      }
    }
  }
}

// ---------------- fused MHSA attention: softmax(QK^T/sqrt(dk)) V ----------------
#define MHSA_LDS (65536 + 3 * 8192 + 32768)
__global__ __launch_bounds__(256) void mhsa_attn(
    const __bf16* __restrict__ Q, const __bf16* __restrict__ Km,
    const __bf16* __restrict__ V, __bf16* __restrict__ Ob) {
  extern __shared__ unsigned char smraw[];
  float*  Sf = (float*)smraw;                          // 128x128 f32
  __bf16* Vt = (__bf16*)(smraw + 65536);               // [d][l] 32x128
  __bf16* Qb = (__bf16*)(smraw + 65536 + 8192);        // [l][d] 128x32
  __bf16* Kb = (__bf16*)(smraw + 65536 + 16384);       // [l][d] 128x32
  __bf16* Sb = (__bf16*)(smraw + 65536 + 24576);       // 128x128 bf16
  int bh = blockIdx.x, head = bh & (NHEAD - 1), bn = bh >> 3;
  int tid = threadIdx.x, lane = tid & 31, wave = tid >> 5;
  const size_t base = (size_t)bn * SEQL * HDIM + head * DK;
  for (int i = tid; i < SEQL * DK; i += 256) {
    int l = i >> 5, d = i & 31;
    size_t gi = base + (size_t)l * HDIM + d;
    Qb[l * 32 + d] = Q[gi];
    Kb[l * 32 + d] = Km[gi];
    Vt[d * 128 + l] = V[gi];
  }
  __syncthreads();
  {
    v16bf a, b;
    load_frag_a(Qb, 32, wave << 4, lane, a);
    const float scale = 0.17677669529663687f;  // 1/sqrt(32)
#pragma unroll
    for (int ct = 0; ct < 8; ++ct) {
      v8f acc = {};
      load_frag_b(Kb, 32, ct << 4, lane, b);
      acc = WMMA_BF16(a, b, acc);
      int col = (ct << 4) + (lane & 15);
      int rb = (wave << 4) + ((lane >> 4) << 3);
#pragma unroll
      for (int r = 0; r < 8; ++r) Sf[(rb + r) * 128 + col] = acc[r] * scale;
    }
  }
  __syncthreads();
  if (tid < 128) {
    float* row = Sf + tid * 128;
    float mx = -1e30f;
    for (int j = 0; j < 128; ++j) mx = fmaxf(mx, row[j]);
    float s = 0.f;
    for (int j = 0; j < 128; ++j) { float e = __expf(row[j] - mx); row[j] = e; s += e; }
    float inv = 1.f / s;
    __bf16* prow = Sb + tid * 128;
    for (int j = 0; j < 128; ++j) prow[j] = (__bf16)(row[j] * inv);
  }
  __syncthreads();
#pragma unroll
  for (int nt = 0; nt < 2; ++nt) {
    v8f acc = {};
    for (int k0 = 0; k0 < 128; k0 += 32) {
      v16bf a, b;
      load_frag_a(Sb + k0, 128, wave << 4, lane, a);
      load_frag_b(Vt + k0, 128, nt << 4, lane, b);
      acc = WMMA_BF16(a, b, acc);
    }
    int col = (nt << 4) + (lane & 15);
    int rb = (wave << 4) + ((lane >> 4) << 3);
#pragma unroll
    for (int r = 0; r < 8; ++r)
      Ob[(size_t)bn * SEQL * HDIM + (size_t)(rb + r) * HDIM + head * DK + col] = (__bf16)acc[r];
  }
}

// ---------------- GAT per-graph attention (padded 62 -> 64), WMMA aggregation ----------------
__global__ __launch_bounds__(256) void gat_attn(
    const __bf16* __restrict__ hNb, const float* __restrict__ f1,
    const float* __restrict__ f2, const int* __restrict__ adj,
    float* __restrict__ outX) {
  __shared__ float  eS[64 * 64];
  __shared__ __bf16 attnS[64 * 64];
  __shared__ __bf16 hT[32 * 64];
  __shared__ float  f1s[64], f2s[64];
  int g = blockIdx.x, b = g >> 7, l = g & (SEQL - 1);
  int tid = threadIdx.x, lane = tid & 31, wave = tid >> 5;
  const int* adjb = adj + (size_t)b * NNODE * NNODE;
  for (int h = 0; h < NHEAD; ++h) {
    if (tid < 64) {
      f1s[tid] = (tid < NNODE) ? f1[((size_t)g * NHEAD + h) * 64 + tid] : 0.f;
      f2s[tid] = (tid < NNODE) ? f2[((size_t)g * NHEAD + h) * 64 + tid] : 0.f;
    }
    __syncthreads();
    for (int i = tid; i < 4096; i += 256) {
      int r = i >> 6, c = i & 63;
      float e = -9e15f;
      if (r < NNODE && c < NNODE && adjb[r * NNODE + c] > 0) {
        float t = f2s[r] + f1s[c];
        e = t > 0.f ? t : 0.2f * t;            // LeakyReLU(0.2)
      }
      eS[i] = e;
    }
    for (int i = tid; i < 2048; i += 256) {
      int d = i >> 6, j = i & 63;
      hT[i] = (j < NNODE) ? hNb[((size_t)g * NNODE + j) * HDIM + h * DK + d] : (__bf16)0.f;
    }
    __syncthreads();
    if (tid < 64) {
      int j = tid;
      float mx = -1e30f;
      for (int i2 = 0; i2 < NNODE; ++i2) mx = fmaxf(mx, eS[i2 * 64 + j]);
      float s = 0.f;
      for (int i2 = 0; i2 < NNODE; ++i2) { float e = __expf(eS[i2 * 64 + j] - mx); eS[i2 * 64 + j] = e; s += e; }
      float inv = 1.f / s;
      for (int i2 = 0; i2 < 64; ++i2)
        attnS[i2 * 64 + j] = (j < NNODE && i2 < NNODE) ? (__bf16)(eS[i2 * 64 + j] * inv) : (__bf16)0.f;
    }
    __syncthreads();
    int mt = (wave >> 1) << 4, nt = (wave & 1) << 4;
    v8f acc = {};
    for (int k0 = 0; k0 < 64; k0 += 32) {
      v16bf a, bb;
      load_frag_a(attnS + k0, 64, mt, lane, a);
      load_frag_b(hT + k0, 64, nt, lane, bb);
      acc = WMMA_BF16(a, bb, acc);
    }
    int node0 = mt + ((lane >> 4) << 3);
    int col = h * DK + nt + (lane & 15);
#pragma unroll
    for (int r = 0; r < 8; ++r) {
      int n = node0 + r;
      if (n < NNODE) {
        float v = acc[r];
        v = v > 0.f ? v : (__expf(v) - 1.f);   // ELU
        outX[(((size_t)b * NNODE + n) * SEQL + l) * HDIM + col] = v;
      }
    }
    __syncthreads();
  }
}

// ---------------- elementwise / reduction kernels ----------------
__global__ void f2b_k(const float* __restrict__ a, __bf16* __restrict__ o, size_t n) {
  size_t i = (size_t)blockIdx.x * 256 + threadIdx.x;
  if (i < n) o[i] = (__bf16)a[i];
}

__global__ void bn_fwd(const float* __restrict__ x, const float* __restrict__ g,
                       const float* __restrict__ bb, const float* __restrict__ rm,
                       const float* __restrict__ rv, float* __restrict__ y, size_t total) {
  size_t i = (size_t)blockIdx.x * 256 + threadIdx.x;
  if (i >= total) return;
  int n = (int)((i / ((size_t)SEQL * HDIM)) % NNODE);
  float inv = g[n] * rsqrtf(rv[n] + 1e-5f);
  y[i] = (x[i] - rm[n]) * inv + bb[n];
}

__global__ __launch_bounds__(256) void ln_bf16(const float* __restrict__ x,
                                               const float* __restrict__ g,
                                               const float* __restrict__ bb,
                                               __bf16* __restrict__ y) {
  __shared__ float red[256];
  int row = blockIdx.x, tid = threadIdx.x;
  float v = x[(size_t)row * HDIM + tid];
  red[tid] = v; __syncthreads();
  for (int s = 128; s > 0; s >>= 1) { if (tid < s) red[tid] += red[tid + s]; __syncthreads(); }
  float m = red[0] * (1.f / HDIM);
  __syncthreads();
  float d = v - m;
  red[tid] = d * d; __syncthreads();
  for (int s = 128; s > 0; s >>= 1) { if (tid < s) red[tid] += red[tid + s]; __syncthreads(); }
  float sd = sqrtf(red[0] * (1.f / (HDIM - 1)));       // unbiased (torch.std)
  y[(size_t)row * HDIM + tid] = (__bf16)(g[tid] * d / (sd + 1e-6f) + bb[tid]);
}

__global__ void add_k(const float* __restrict__ a, const float* __restrict__ b,
                      float* __restrict__ c, size_t n) {
  size_t i = (size_t)blockIdx.x * 256 + threadIdx.x;
  if (i < n) c[i] = a[i] + b[i];
}

__global__ void transpose_nl(const float* __restrict__ x, __bf16* __restrict__ y) {
  size_t i = (size_t)blockIdx.x * 256 + threadIdx.x;
  if (i >= EL) return;
  int hc = (int)(i & (HDIM - 1));
  size_t r = i >> 8;
  int n = (int)(r % NNODE); r /= NNODE;
  int l = (int)(r % SEQL);
  int b = (int)(r / SEQL);
  y[i] = (__bf16)x[(((size_t)b * NNODE + n) * SEQL + l) * HDIM + hc];
}

__global__ void pack_gat_w(const float* __restrict__ w, __bf16* __restrict__ wb) {
  int idx = blockIdx.x * 256 + threadIdx.x;
  if (idx >= HDIM * HDIM) return;
  int col = idx & (HDIM - 1), f = idx >> 8;
  int h = col >> 5, d = col & 31;
  wb[idx] = (__bf16)w[((size_t)h * HDIM + f) * DK + d];
}

__global__ void gat_scores(const float* __restrict__ hN, const float* __restrict__ ga,
                           float* __restrict__ f1, float* __restrict__ f2) {
  int idx = blockIdx.x * 256 + threadIdx.x;
  if (idx >= NGRAPH * NHEAD * 64) return;
  int n = idx & 63, h = (idx >> 6) & 7, g = idx >> 9;
  float s1 = 0.f, s2 = 0.f;
  if (n < NNODE) {
    const float* hp = hN + ((size_t)g * NNODE + n) * HDIM + h * DK;
    const float* a1 = ga + h * 2 * DK;
    const float* a2 = a1 + DK;
#pragma unroll
    for (int d = 0; d < DK; ++d) { float v = hp[d]; s1 += v * a1[d]; s2 += v * a2[d]; }
  }
  f1[idx] = s1; f2[idx] = s2;
}

__global__ __launch_bounds__(256) void sum_nl(const float* __restrict__ x, float* __restrict__ out) {
  __shared__ float red[256];
  int b = blockIdx.x >> 8, hc = blockIdx.x & (HDIM - 1);
  int tid = threadIdx.x;
  float s = 0.f;
  for (int i = tid; i < NNODE * SEQL; i += 256)
    s += x[((size_t)b * NNODE * SEQL + i) * HDIM + hc];
  red[tid] = s; __syncthreads();
  for (int sh = 128; sh > 0; sh >>= 1) { if (tid < sh) red[tid] += red[tid + sh]; __syncthreads(); }
  if (tid == 0) out[b * HDIM + hc] = red[0];
}

__global__ __launch_bounds__(256) void head_k(const float* __restrict__ s,
                                              const float* __restrict__ lw, const float* __restrict__ lb,
                                              const float* __restrict__ ow, const float* __restrict__ ob,
                                              float* __restrict__ out) {
  __shared__ float hid[HDIM];
  __shared__ float logits[3];
  int tid = threadIdx.x;
  for (int b = 0; b < BATCH; ++b) {
    float acc = lb[tid];
    for (int k = 0; k < HDIM; ++k) acc += s[b * HDIM + k] * lw[k * HDIM + tid];
    hid[tid] = tanhf(acc);
    __syncthreads();
    if (tid < 3) {
      float a2 = ob[tid];
      for (int j = 0; j < HDIM; ++j) a2 += hid[j] * ow[j * 3 + tid];
      logits[tid] = a2;
    }
    __syncthreads();
    if (tid == 0) {
      float mx = fmaxf(logits[0], fmaxf(logits[1], logits[2]));
      float sum = __expf(logits[0] - mx) + __expf(logits[1] - mx) + __expf(logits[2] - mx);
      float lse = mx + logf(sum);
      out[b * 3 + 0] = logits[0] - lse;
      out[b * 3 + 1] = logits[1] - lse;
      out[b * 3 + 2] = logits[2] - lse;
    }
    __syncthreads();
  }
}

// ---------------- orchestration ----------------
extern "C" void kernel_launch(void* const* d_in, const int* in_sizes, int n_in,
                              void* d_out, int out_size, void* d_ws, size_t ws_size,
                              hipStream_t stream) {
  (void)in_sizes; (void)n_in; (void)out_size; (void)ws_size;
  const float* x     = (const float*)d_in[0];
  const int*   adj   = (const int*)d_in[1];
  const float* mlp_w = (const float*)d_in[2];
  const float* mlp_b = (const float*)d_in[3];
  const float* lin_w = (const float*)d_in[4];
  const float* lin_b = (const float*)d_in[5];
  const float* out_w = (const float*)d_in[6];
  const float* out_b = (const float*)d_in[7];

  char* wp = (char*)d_ws;
  auto carve = [&](size_t bytes) { char* p = wp; wp += (bytes + 255) & ~(size_t)255; return p; };

  __bf16* xb_in  = (__bf16*)carve((size_t)M1 * FIN * 2);
  __bf16* wb_mlp = (__bf16*)carve((size_t)FIN * HDIM * 2);
  __bf16* wb_q[2], *wb_k[2], *wb_v[2], *wb_o[2], *wb_g[2];
  for (int l = 0; l < 2; ++l) {
    wb_q[l] = (__bf16*)carve((size_t)HDIM * HDIM * 2);
    wb_k[l] = (__bf16*)carve((size_t)HDIM * HDIM * 2);
    wb_v[l] = (__bf16*)carve((size_t)HDIM * HDIM * 2);
    wb_o[l] = (__bf16*)carve((size_t)HDIM * HDIM * 2);
    wb_g[l] = (__bf16*)carve((size_t)HDIM * HDIM * 2);
  }
  float*  bufX  = (float*)carve(EL * 4);
  float*  bufA  = (float*)carve(EL * 4);
  float*  bufB  = (float*)carve(EL * 4);
  __bf16* bufLb = (__bf16*)carve(EL * 2);
  __bf16* qb    = (__bf16*)carve(EL * 2);
  __bf16* kb    = (__bf16*)carve(EL * 2);
  __bf16* vb    = (__bf16*)carve(EL * 2);
  __bf16* obuf  = (__bf16*)carve(EL * 2);
  float*  f1    = (float*)carve((size_t)NGRAPH * NHEAD * 64 * 4);
  float*  f2    = (float*)carve((size_t)NGRAPH * NHEAD * 64 * 4);
  float*  sumb  = (float*)carve((size_t)BATCH * HDIM * 4);

  const int T = 256;
  auto grid1 = [&](size_t n) { return dim3((unsigned)((n + T - 1) / T)); };

  f2b_k<<<grid1((size_t)M1 * FIN), T, 0, stream>>>(x, xb_in, (size_t)M1 * FIN);
  f2b_k<<<grid1((size_t)FIN * HDIM), T, 0, stream>>>(mlp_w, wb_mlp, (size_t)FIN * HDIM);
  for (int l = 0; l < 2; ++l) {
    const int base = 8 + 16 * l;
    f2b_k<<<grid1(65536), T, 0, stream>>>((const float*)d_in[base + 0], wb_q[l], 65536);
    f2b_k<<<grid1(65536), T, 0, stream>>>((const float*)d_in[base + 2], wb_k[l], 65536);
    f2b_k<<<grid1(65536), T, 0, stream>>>((const float*)d_in[base + 4], wb_v[l], 65536);
    f2b_k<<<grid1(65536), T, 0, stream>>>((const float*)d_in[base + 6], wb_o[l], 65536);
    pack_gat_w<<<grid1(65536), T, 0, stream>>>((const float*)d_in[base + 14], wb_g[l]);
  }

  // input MLP: (M1,128) @ (128,256) + b
  dim3 gg(HDIM / 128, M1 / 64);
  gemm_bf16<<<gg, T, GEMM_LDS, stream>>>(xb_in, wb_mlp, mlp_b, bufX, nullptr, M1, HDIM, FIN);

  for (int l = 0; l < 2; ++l) {
    const int base = 8 + 16 * l;
    const float* bq    = (const float*)d_in[base + 1];
    const float* bk    = (const float*)d_in[base + 3];
    const float* bv    = (const float*)d_in[base + 5];
    const float* bo    = (const float*)d_in[base + 7];
    const float* ln_g  = (const float*)d_in[base + 8];
    const float* ln_b  = (const float*)d_in[base + 9];
    const float* bn_g  = (const float*)d_in[base + 10];
    const float* bn_b  = (const float*)d_in[base + 11];
    const float* bn_rm = (const float*)d_in[base + 12];
    const float* bn_rv = (const float*)d_in[base + 13];
    const float* gat_a = (const float*)d_in[base + 15];

    bn_fwd<<<grid1(EL), T, 0, stream>>>(bufX, bn_g, bn_b, bn_rm, bn_rv, bufA, EL);
    ln_bf16<<<M1, T, 0, stream>>>(bufA, ln_g, ln_b, bufLb);

    gemm_bf16<<<gg, T, GEMM_LDS, stream>>>(bufLb, wb_q[l], bq, nullptr, qb, M1, HDIM, HDIM);
    gemm_bf16<<<gg, T, GEMM_LDS, stream>>>(bufLb, wb_k[l], bk, nullptr, kb, M1, HDIM, HDIM);
    gemm_bf16<<<gg, T, GEMM_LDS, stream>>>(bufLb, wb_v[l], bv, nullptr, vb, M1, HDIM, HDIM);

    mhsa_attn<<<BATCH * NNODE * NHEAD, T, MHSA_LDS, stream>>>(qb, kb, vb, obuf);

    gemm_bf16<<<gg, T, GEMM_LDS, stream>>>(obuf, wb_o[l], bo, bufB, nullptr, M1, HDIM, HDIM);
    add_k<<<grid1(EL), T, 0, stream>>>(bufA, bufB, bufX, EL);           // residual
    bn_fwd<<<grid1(EL), T, 0, stream>>>(bufX, bn_g, bn_b, bn_rm, bn_rv, bufA, EL);
    transpose_nl<<<grid1(EL), T, 0, stream>>>(bufA, bufLb);             // (b,l,n,h) bf16

    gemm_bf16<<<gg, T, GEMM_LDS, stream>>>(bufLb, wb_g[l], nullptr, bufB, qb, M1, HDIM, HDIM);
    gat_scores<<<grid1((size_t)NGRAPH * NHEAD * 64), T, 0, stream>>>(bufB, gat_a, f1, f2);
    gat_attn<<<NGRAPH, T, 0, stream>>>(qb, f1, f2, adj, bufX);          // writes (b,n,l,h)
  }

  sum_nl<<<BATCH * HDIM, T, 0, stream>>>(bufX, sumb);
  head_k<<<1, T, 0, stream>>>(sumb, lin_w, lin_b, out_w, out_b, (float*)d_out);
}
